// PathIntegration_86337432584803
// MI455X (gfx1250) — compile-verified, matrix operand order checked by reading, not run on hardware
//
#include <hip/hip_runtime.h>
#include <hip/hip_bf16.h>

// Problem constants (from reference): B=32, T=512, A=64, D=128, H=256
#define BB   32
#define TT   512
#define AA   64
#define DD   128
#define HH   256
#define BT   (BB*TT)        // 16384
#define KTOT (HH*DD)        // 32768 combined (k,i) index
#define NWG  64             // persistent workgroups for recurrence
#define KC   (KTOT/NWG)     // 512 K-chunk per WG

typedef __attribute__((ext_vector_type(16))) __bf16 v16bf;
typedef __attribute__((ext_vector_type(8)))  float  v8f;
typedef unsigned int u32x4 __attribute__((ext_vector_type(4)));
typedef int          i32x8 __attribute__((ext_vector_type(8)));
typedef int          i32x4 __attribute__((ext_vector_type(4)));

union Frag16 { unsigned int u[8]; v16bf v; };

__device__ __forceinline__ unsigned short f32_bf16(float f) {
  unsigned int x = __float_as_uint(f);
  unsigned int r = x + 0x7FFFu + ((x >> 16) & 1u);   // round-to-nearest-even
  return (unsigned short)(r >> 16);
}
__device__ __forceinline__ float bf16_f32(unsigned short h) {
  return __uint_as_float(((unsigned int)h) << 16);
}
__device__ __forceinline__ unsigned int pack2(unsigned short lo, unsigned short hi) {
  return (unsigned int)lo | ((unsigned int)hi << 16);
}
// K index base inside a 32-wide K block for bf16 A/B fragments (ISA 7.12.2):
// lanes 0-15 hold K {0..7,16..23}, lanes 16-31 hold K {8..15,24..31}, packed 2/VGPR.
__device__ __forceinline__ int kbase(int v, int half) {
  return (v < 4 ? 2 * v : 16 + 2 * (v - 4)) + 8 * half;
}

// ---- TDM: 1-D contiguous 8 KB tile (32 x 128 bf16 w2 block) -> LDS ---------
// D# per CDNA5 ISA: group0 {count=1, lds_addr,
// global_addr, type=2}; group1 {data_size=2B, tensor_dim0=tile_dim0=4096,
// tensor_dim1=tile_dim1=1}; groups 2/3 zero (<=2D tensor).
// This toolchain's builtin takes 6 args (extra i32x8 group before cpol).
__device__ __forceinline__ void tdm_load_8k(const unsigned short* gsrc,
                                            unsigned int lds_off) {
  unsigned long long ga = (unsigned long long)(uintptr_t)gsrc;
  u32x4 g0;
  g0[0] = 1u;                                             // count=1 (valid)
  g0[1] = lds_off;                                        // lds_addr (bytes)
  g0[2] = (unsigned int)(ga & 0xFFFFFFFFull);             // global_addr[31:0]
  g0[3] = (unsigned int)((ga >> 32) & 0x1FFFFFFull)       // global_addr[56:32]
        | (2u << 30);                                     // type=2 ("image")
  i32x8 g1;
  g1[0] = (int)(1u << 16);        // data_size=1 -> 2 bytes/element
  g1[1] = (int)(4096u << 16);     // tensor_dim0[15:0]=4096
  g1[2] = (int)(1u << 16);        // tensor_dim0[31:16]=0, tensor_dim1[15:0]=1
  g1[3] = (int)(4096u << 16);     // tensor_dim1[31:16]=0, tile_dim0=4096
  g1[4] = 1;                      // tile_dim1=1, tile_dim2=0
  g1[5] = 4096;                   // tensor_dim0_stride[31:0]
  g1[6] = 0;
  g1[7] = 0;
  i32x4 gz4 = {0, 0, 0, 0};
  i32x8 gz8 = {0, 0, 0, 0, 0, 0, 0, 0};
  __builtin_amdgcn_tensor_load_to_lds(g0, g1, gz4, gz4, gz8, 0);
}

// ---------------- prep: cast inputs to bf16, reset barrier counter ----------
__global__ void k_cast(const float* __restrict__ actions,
                       const float* __restrict__ w1,
                       const float* __restrict__ w2,
                       unsigned short* __restrict__ a_bf,
                       unsigned short* __restrict__ w1_bf,
                       unsigned short* __restrict__ w2_bf,
                       unsigned int* counter) {
  long i = (long)blockIdx.x * blockDim.x + threadIdx.x;
  long stride = (long)gridDim.x * blockDim.x;
  if (i == 0) *counter = 0u;
  for (long k = i; k < (long)BT * AA; k += stride) a_bf[k] = f32_bf16(actions[k]);
  for (long k = i; k < (long)AA * HH; k += stride) w1_bf[k] = f32_bf16(w1[k]);
  for (long k = i; k < (long)HH * DD * DD; k += stride) w2_bf[k] = f32_bf16(w2[k]);
}

// ---------------- s0 = l2norm(init_structure), broadcast over batch ---------
__global__ void k_init_s(const float* __restrict__ init_structure,
                         float* __restrict__ s_cur) {
  __shared__ float red[DD];
  int d = threadIdx.x;
  float x = init_structure[d];
  red[d] = x * x;
  __syncthreads();
  for (int off = DD / 2; off > 0; off >>= 1) {
    if (d < off) red[d] += red[d + off];
    __syncthreads();
  }
  float s = x / fmaxf(sqrtf(red[0]), 1e-12f);
  for (int b = 0; b < BB; ++b) s_cur[b * DD + d] = s;
}

// ---------------- h = relu(actions @ w1 + b1), bf16 out (WMMA) --------------
__global__ void __launch_bounds__(256) k_mlp1(const unsigned short* __restrict__ a_bf,
                                              const unsigned short* __restrict__ w1_bf,
                                              const float* __restrict__ b1,
                                              unsigned short* __restrict__ h_bf) {
  int wave = (int)((blockIdx.x * blockDim.x + threadIdx.x) >> 5);
  int lane = threadIdx.x & 31;
  int rt = wave >> 4;           // 0..1023 row tiles of 16 (over B*T)
  int ct = wave & 15;           // 0..15 col tiles of 16 (over H)
  int half = lane >> 4;
  int mn = lane & 15;
  int row0 = rt * 16, col0 = ct * 16;
  v8f c = {};
  #pragma unroll
  for (int kb = 0; kb < AA; kb += 32) {
    Frag16 a, b;
    #pragma unroll
    for (int v = 0; v < 8; ++v) {
      int k = kb + kbase(v, half);
      const unsigned short* ap = a_bf + (long)(row0 + mn) * AA + k;  // K consecutive
      a.u[v] = pack2(ap[0], ap[1]);
      b.u[v] = pack2(w1_bf[(long)k * HH + col0 + mn],                // w1[k][n]
                     w1_bf[(long)(k + 1) * HH + col0 + mn]);
    }
    c = __builtin_amdgcn_wmma_f32_16x16x32_bf16(false, a.v, false, b.v, (short)0, c,
                                                false, false);
  }
  #pragma unroll
  for (int v = 0; v < 8; ++v) {
    int m = v + 8 * half;
    int col = col0 + mn;
    float val = fmaxf(c[v] + b1[col], 0.0f);
    h_bf[(long)(row0 + m) * HH + col] = f32_bf16(val);
  }
}

// ---------------- persistent fused recurrence -------------------------------
__device__ __forceinline__ void grid_barrier(unsigned int* counter, unsigned int target) {
  __syncthreads();
  if (threadIdx.x == 0) {
    __threadfence();
    atomicAdd(counter, 1u);
    while (atomicAdd(counter, 0u) < target) { __builtin_amdgcn_s_sleep(2); }
  }
  __syncthreads();
  __threadfence();
}

__global__ void __launch_bounds__(256) k_recur(const unsigned short* __restrict__ h_bf,
                                               const unsigned short* __restrict__ w2_bf,
                                               const float* __restrict__ b2,
                                               float* s_cur, float* partials,
                                               unsigned int* counter,
                                               float* __restrict__ ys) {
  __shared__ float sh_s[BB * DD];                              // 16 KB state s_t
  __shared__ unsigned short sh_h[BB * HH];                     // 16 KB h[:, t, :]
  __shared__ __align__(16) unsigned short sh_w2[2][32 * DD];   // 2 x 8 KB w2 blocks (TDM dest)
  __shared__ __align__(8)  unsigned short sh_a[2][32 * 32];    // 2 x 2 KB A = outer(h,s) tiles
  __shared__ float sh_red[DD];

  const int tid = threadIdx.x;
  const int wg = blockIdx.x;          // 0..63, K-slice owner
  const int lane = tid & 31;
  const int ww = tid >> 5;            // wave 0..7 -> N strip
  const int half = lane >> 4;
  const int mn = lane & 15;
  const int j0 = ww * 16;
  const long k0 = (long)wg * KC;
  const unsigned int w2_lds[2] = { (unsigned int)(uintptr_t)&sh_w2[0][0],
                                   (unsigned int)(uintptr_t)&sh_w2[1][0] };
  // cooperative A-tile build: 1024 bf16 elems / 256 threads = 4 each (same row)
  const int ar  = tid >> 3;           // row 0..31
  const int akk = (tid & 7) * 4;      // K offset within 32-block, 4-aligned

  for (int t = 0; t < TT; ++t) {
    // Stage s_t and h[:, t, :] into LDS (coalesced)
    for (int i = tid; i < BB * DD; i += 256) sh_s[i] = s_cur[i];
    for (int i = tid; i < BB * HH; i += 256) {
      int b = i >> 8, k = i & (HH - 1);
      sh_h[i] = h_bf[((long)b * TT + t) * HH + k];
    }
    __syncthreads();   // sh_s/sh_h ready for A build

    // Preload pipeline stage 0: TDM w2 block 0 + build A block 0
    if (tid < 32) tdm_load_8k(w2_bf + k0 * DD, w2_lds[0]);
    {
      long ki = k0 + akk;
      int kcol = (int)(ki >> 7), i0 = (int)(ki & (DD - 1));
      float hv = bf16_f32(sh_h[ar * HH + kcol]);
      unsigned int* dst = (unsigned int*)&sh_a[0][ar * 32 + akk];
      dst[0] = pack2(f32_bf16(hv * sh_s[ar * DD + i0]),
                     f32_bf16(hv * sh_s[ar * DD + i0 + 1]));
      dst[1] = pack2(f32_bf16(hv * sh_s[ar * DD + i0 + 2]),
                     f32_bf16(hv * sh_s[ar * DD + i0 + 3]));
    }
    if (tid < 32) __builtin_amdgcn_s_wait_tensorcnt(0);
    __syncthreads();

    v8f c0 = {}, c1 = {};
    #pragma unroll 1
    for (int kb = 0; kb < KC; kb += 32) {
      int buf = (kb >> 5) & 1;
      if (kb + 32 < KC) {
        // async DMA of next w2 block while we compute this one
        if (tid < 32) tdm_load_8k(w2_bf + (k0 + kb + 32) * DD, w2_lds[buf ^ 1]);
        // build next A tile cooperatively (overlaps with WMMA below)
        long ki = k0 + kb + 32 + akk;
        int kcol = (int)(ki >> 7), i0 = (int)(ki & (DD - 1));
        float hv = bf16_f32(sh_h[ar * HH + kcol]);
        unsigned int* dst = (unsigned int*)&sh_a[buf ^ 1][ar * 32 + akk];
        dst[0] = pack2(f32_bf16(hv * sh_s[ar * DD + i0]),
                       f32_bf16(hv * sh_s[ar * DD + i0 + 1]));
        dst[1] = pack2(f32_bf16(hv * sh_s[ar * DD + i0 + 2]),
                       f32_bf16(hv * sh_s[ar * DD + i0 + 3]));
      }
      if (kb + 64 < KC)
        __builtin_prefetch(w2_bf + (k0 + kb + 64) * DD, 0, 1);

      // Fragments: A tiles are pre-packed in LDS (plain b32 loads), B packed here
      Frag16 a0, a1, bf;
      #pragma unroll
      for (int v = 0; v < 8; ++v) {
        int kk = kbase(v, half);
        a0.u[v] = *(const unsigned int*)&sh_a[buf][mn * 32 + kk];
        a1.u[v] = *(const unsigned int*)&sh_a[buf][(16 + mn) * 32 + kk];
        bf.u[v] = pack2(sh_w2[buf][kk * DD + j0 + mn],
                        sh_w2[buf][(kk + 1) * DD + j0 + mn]);
      }
      c0 = __builtin_amdgcn_wmma_f32_16x16x32_bf16(false, a0.v, false, bf.v, (short)0,
                                                   c0, false, false);
      c1 = __builtin_amdgcn_wmma_f32_16x16x32_bf16(false, a1.v, false, bf.v, (short)0,
                                                   c1, false, false);
      if (tid < 32) __builtin_amdgcn_s_wait_tensorcnt(0);  // DMA done (wave 0)
      __syncthreads();  // publish next buffers; current fully consumed
    }

    // Write this WG's deterministic partial tile [32 x 128]
    float* pslot = partials + (long)wg * (BB * DD);
    #pragma unroll
    for (int v = 0; v < 8; ++v) {
      int m = v + 8 * half;
      pslot[m * DD + j0 + mn]        = c0[v];
      pslot[(16 + m) * DD + j0 + mn] = c1[v];
    }
    __threadfence();
    unsigned int base = (unsigned int)(2 * t) * NWG;
    grid_barrier(counter, base + NWG);

    // Phase 2: WG b (<32) reduces K-slices in fixed order, bias, relu, l2norm
    if (wg < BB) {
      const int b = wg, j = tid;
      float acc = 0.f;
      if (tid < DD) {
        for (int sl = 0; sl < NWG; ++sl)
          acc += partials[(long)sl * (BB * DD) + b * DD + j];
        float bias = 0.f;
        for (int i2 = 0; i2 < DD; ++i2)
          bias = fmaf(sh_s[b * DD + i2], b2[i2 * DD + j], bias);  // s . b2 column
        acc = fmaxf(acc + bias, 0.f);
        sh_red[j] = acc * acc;
      }
      __syncthreads();
      for (int off = DD / 2; off > 0; off >>= 1) {
        if (tid < off) sh_red[tid] += sh_red[tid + off];
        __syncthreads();
      }
      float denom = fmaxf(sqrtf(sh_red[0]), 1e-12f);
      if (tid < DD) {
        float outv = acc / denom;
        ys[((long)b * TT + t) * DD + j] = outv;   // output [B,T,D]
        s_cur[b * DD + j] = outv;
      }
    }
    __threadfence();
    grid_barrier(counter, base + 2 * NWG);
  }
}

// ---------------------------------------------------------------------------
extern "C" void kernel_launch(void* const* d_in, const int* in_sizes, int n_in,
                              void* d_out, int out_size, void* d_ws, size_t ws_size,
                              hipStream_t stream) {
  const float* actions = (const float*)d_in[0];   // [32,512,64]
  const float* init_s  = (const float*)d_in[1];   // [128]
  const float* w1      = (const float*)d_in[2];   // [64,256]
  const float* b1      = (const float*)d_in[3];   // [256]
  const float* w2      = (const float*)d_in[4];   // [256,16384]
  const float* b2      = (const float*)d_in[5];   // [16384]
  float* ys = (float*)d_out;                      // [32,512,128]

  char* ws = (char*)d_ws;
  unsigned short* h_bf  = (unsigned short*)(ws);               // 8,388,608 B
  unsigned short* a_bf  = (unsigned short*)(ws + 8388608);     // 2,097,152 B
  unsigned short* w1_bf = (unsigned short*)(ws + 10485760);    //    32,768 B
  unsigned short* w2_bf = (unsigned short*)(ws + 10518528);    // 8,388,608 B
  float* s_cur          = (float*)(ws + 18907136);             //    16,384 B
  float* partials       = (float*)(ws + 18923520);             // 1,048,576 B
  unsigned int* counter = (unsigned int*)(ws + 19972096);      //        64 B

  k_cast  <<<1024, 256, 0, stream>>>(actions, w1, w2, a_bf, w1_bf, w2_bf, counter);
  k_init_s<<<1,    128, 0, stream>>>(init_s, s_cur);
  k_mlp1  <<<2048, 256, 0, stream>>>(a_bf, w1_bf, b1, h_bf);
  k_recur <<<NWG,  256, 0, stream>>>(h_bf, w2_bf, b2, s_cur, partials, counter, ys);
}
